// GraphLSTM_26250840113392
// MI455X (gfx1250) — matrix operand order repaired and checked by reading, not compiled, not run on hardware
//
#include <hip/hip_runtime.h>
#include <hip/hip_bf16.h>

// ---------------------------------------------------------------------------
// GraphLSTM for MI455X (gfx1250), wave32 + v_wmma_f32_16x16x32_bf16.
// S=20, N=4096, IN=2, DE=GH=MH=128, CH=256, OUT=5, CE=384.
// ---------------------------------------------------------------------------

typedef __attribute__((ext_vector_type(16))) __bf16 v16bf;
typedef __attribute__((ext_vector_type(8)))  float  v8f;

#define SEQ   20
#define NN    4096
#define WMMA_BF16(a, b, c) \
    __builtin_amdgcn_wmma_f32_16x16x32_bf16(false, (a), false, (b), (short)0, (c), false, false)

__device__ __forceinline__ float sigm(float x) { return 1.f / (1.f + __expf(-x)); }

// A fragment: 16x32 bf16 tile, row-major source. Lane L: row = L%16,
// elements 0..7 = K[koff..koff+7], 8..15 = K[koff+16..koff+23], koff = (L<16)?0:8.
__device__ __forceinline__ v16bf load_a_frag(const __bf16* rowk, int lane) {
    const int koff = (lane & 16) ? 8 : 0;
    union U { v16bf v; uint4 q[2]; } u;
    u.q[0] = *reinterpret_cast<const uint4*>(rowk + koff);
    u.q[1] = *reinterpret_cast<const uint4*>(rowk + koff + 16);
    return u.v;
}

// ---------------------------------------------------------------------------
// Weight prep: pack logical W[k][j] (K x Nout) into WMMA B-fragment order:
// dst[((kc*colTiles + ct)*32 + lane)*16 + e] = W[kc*32 + (lane>=16?8:0) + (e&7) + (e>=8?16:0)][ct*16 + lane%16]
// transA: W[k][j] = srcA[k*ldA + j]  (already K-major, e.g. w_graph_veh)
// else:   W[k][j] = k < Ksplit ? srcA[j*ldA + k] : srcB[j*ldB + k-Ksplit]
// ---------------------------------------------------------------------------
__global__ void pack_b_kernel(__bf16* __restrict__ dst,
                              const float* __restrict__ srcA,
                              const float* __restrict__ srcB,
                              int K, int Ksplit, int Nout, int ldA, int ldB, int transA) {
    const int total = K * Nout;
    const int colTiles = Nout >> 4;
    for (int p = blockIdx.x * blockDim.x + threadIdx.x; p < total;
         p += gridDim.x * blockDim.x) {
        int e    = p & 15;
        int lane = (p >> 4) & 31;
        int rest = p >> 9;
        int ct   = rest % colTiles;
        int kc   = rest / colTiles;
        int k    = kc * 32 + ((lane & 16) ? 8 : 0) + (e & 7) + ((e & 8) ? 16 : 0);
        int j    = ct * 16 + (lane & 15);
        float v;
        if (transA)           v = srcA[k * ldA + j];
        else if (k < Ksplit)  v = srcA[j * ldA + k];
        else                  v = srcB[j * ldB + (k - Ksplit)];
        dst[p] = (__bf16)v;
    }
}

__global__ void bias_kernel(const float* bihV, const float* bhhV,
                            const float* bihP, const float* bhhP,
                            float* bV, float* bP) {
    int t = blockIdx.x * blockDim.x + threadIdx.x;
    if (t < 512) { bV[t] = bihV[t] + bhhV[t]; bP[t] = bihP[t] + bhhP[t]; }
}

// Copy initial recurrent state into f32 working buffers + bf16 A-panels.
__global__ void init_kernel(const float* cell_h, const float* cell_c,
                            const float* gveh_h, const float* gveh_c,
                            const float* gped_h, const float* gped_c,
                            float* ch, float* cc, float* gvh, float* gvc,
                            float* gph, float* gpc,
                            __bf16* cellAct, __bf16* actV, __bf16* actP) {
    for (int t = blockIdx.x * blockDim.x + threadIdx.x; t < NN * 256;
         t += gridDim.x * blockDim.x) {
        ch[t] = cell_h[t];
        cc[t] = cell_c[t];
        int n = t >> 8, h = t & 255;
        cellAct[(size_t)n * 512 + 256 + h] = (__bf16)cell_h[t];
        if (t < NN * 128) {
            gvh[t] = gveh_h[t]; gvc[t] = gveh_c[t];
            gph[t] = gped_h[t]; gpc[t] = gped_c[t];
            int n2 = t >> 7, h2 = t & 127;
            actV[n2 * 256 + 128 + h2] = (__bf16)gveh_h[t];
            actP[n2 * 256 + 128 + h2] = (__bf16)gped_h[t];
        }
    }
}

// Input embeddings (K=2 -> VALU): dyn -> cellAct[:,0:128], gve/gpe -> actV/actP[:,0:128]
__global__ void embed_kernel(const float* __restrict__ veh, const float* __restrict__ ped,
                             const float* __restrict__ Wd, const float* __restrict__ bd,
                             const float* __restrict__ Wv, const float* __restrict__ bv,
                             const float* __restrict__ Wp, const float* __restrict__ bp,
                             __bf16* cellAct, __bf16* actV, __bf16* actP) {
    for (int t = blockIdx.x * blockDim.x + threadIdx.x; t < NN * 128;
         t += gridDim.x * blockDim.x) {
        int n = t >> 7, j = t & 127;
        float v0 = veh[n * 2], v1 = veh[n * 2 + 1];
        float p0 = ped[n * 2], p1 = ped[n * 2 + 1];
        float d  = fmaxf(0.f, fmaf(v0, Wd[j * 2], fmaf(v1, Wd[j * 2 + 1], bd[j])));
        float gv = fmaxf(0.f, fmaf(v0, Wv[j * 2], fmaf(v1, Wv[j * 2 + 1], bv[j])));
        float gp = fmaxf(0.f, fmaf(p0, Wp[j * 2], fmaf(p1, Wp[j * 2 + 1], bp[j])));
        cellAct[(size_t)n * 512 + j] = (__bf16)d;
        actV[n * 256 + j] = (__bf16)gv;
        actP[n * 256 + j] = (__bf16)gp;
    }
}

// ---------------------------------------------------------------------------
// Graph LSTMs (gveh: blockIdx.y=0, gped: blockIdx.y=1).
// A = [emb | h_prev] bf16 [N,256]; B packed [256 -> 512]; fused gates in LDS.
// Block: 16 rows x 512 cols, 8 waves, each wave 4 16x16 tiles.
// ---------------------------------------------------------------------------
__global__ __launch_bounds__(256) void glstm_kernel(
    __bf16* actV, __bf16* actP,
    const __bf16* __restrict__ WpkV, const __bf16* __restrict__ WpkP,
    const float* __restrict__ bV, const float* __restrict__ bP,
    float* gvh, float* gvc, float* gph, float* gpc) {
    __shared__ float g[16][512];
    const int lane = threadIdx.x & 31;
    const int wave = threadIdx.x >> 5;
    const int row0 = blockIdx.x * 16;

    __bf16* act       = blockIdx.y ? actP : actV;
    const __bf16* Wpk = blockIdx.y ? WpkP : WpkV;
    const float* bias = blockIdx.y ? bP : bV;
    float* H          = blockIdx.y ? gph : gvh;
    float* C          = blockIdx.y ? gpc : gvc;

    const int colBase = wave * 64;
    const int m = lane & 15;
    v8f a0 = {}, a1 = {}, a2 = {}, a3 = {};
#pragma unroll
    for (int kc = 0; kc < 8; ++kc) {
        v16bf a = load_a_frag(act + (size_t)(row0 + m) * 256 + kc * 32, lane);
        const v16bf* bp = reinterpret_cast<const v16bf*>(Wpk) +
                          ((size_t)(kc * 32 + (colBase >> 4)) * 32 + lane);
        a0 = WMMA_BF16(a, bp[0],  a0);
        a1 = WMMA_BF16(a, bp[32], a1);
        a2 = WMMA_BF16(a, bp[64], a2);
        a3 = WMMA_BF16(a, bp[96], a3);
    }
    const int n  = lane & 15;
    const int mh = (lane & 16) ? 8 : 0;
#pragma unroll
    for (int r = 0; r < 8; ++r) {
        g[mh + r][colBase +  0 + n] = a0[r] + bias[colBase +  0 + n];
        g[mh + r][colBase + 16 + n] = a1[r] + bias[colBase + 16 + n];
        g[mh + r][colBase + 32 + n] = a2[r] + bias[colBase + 32 + n];
        g[mh + r][colBase + 48 + n] = a3[r] + bias[colBase + 48 + n];
    }
    __syncthreads();
    // gate order i,f,g,o ; H = 128
    for (int e = threadIdx.x; e < 16 * 128; e += 256) {
        int r = e >> 7, h = e & 127, node = row0 + r;
        float gi = g[r][h], gf = g[r][128 + h], gg = g[r][256 + h], go = g[r][384 + h];
        float cp = C[node * 128 + h];
        float c2 = sigm(gf) * cp + sigm(gi) * tanhf(gg);
        float h2 = sigm(go) * tanhf(c2);
        C[node * 128 + h] = c2;
        H[node * 128 + h] = h2;
        act[(size_t)node * 256 + 128 + h] = (__bf16)h2;  // A-panel for next GEMMs
    }
}

// Column sums of gvh/gph over all N nodes -> sumV/sumP [128]
__global__ __launch_bounds__(256) void reduce_kernel(const float* __restrict__ gvh,
                                                     const float* __restrict__ gph,
                                                     float* sumV, float* sumP) {
    const float* src = blockIdx.y ? gph : gvh;
    float* dst       = blockIdx.y ? sumP : sumV;
    const int c = blockIdx.x;
    float a = 0.f;
    for (int r = threadIdx.x; r < NN; r += 256) a += src[(size_t)r * 128 + c];
    __shared__ float red[256];
    red[threadIdx.x] = a;
    __syncthreads();
    for (int s = 128; s > 0; s >>= 1) {
        if (threadIdx.x < s) red[threadIdx.x] += red[threadIdx.x + s];
        __syncthreads();
    }
    if (threadIdx.x == 0) dst[c] = red[0];
}

// sumVW = (sum gvh) @ w_graph_veh ; cellb = bih+bhh + ((sum gph)@w_graph_ped)@Wih_cell[:,256:384]
__global__ __launch_bounds__(1024) void proj_kernel(
    const float* __restrict__ sumV, const float* __restrict__ sumP,
    const float* __restrict__ wgv,  const float* __restrict__ wgp,
    const float* __restrict__ Wih_cell,
    const float* __restrict__ bih,  const float* __restrict__ bhh,
    float* sumVW, float* cellb) {
    __shared__ float pedRow[128];
    const int t = threadIdx.x;
    if (t < 128) {
        float a = 0.f;
        for (int k = 0; k < 128; ++k) a = fmaf(sumP[k], wgp[k * 128 + t], a);
        pedRow[t] = a;
    } else if (t < 256) {
        int j = t - 128;
        float a = 0.f;
        for (int k = 0; k < 128; ++k) a = fmaf(sumV[k], wgv[k * 128 + j], a);
        sumVW[j] = a;
    }
    __syncthreads();
    float a = bih[t] + bhh[t];
    for (int k = 0; k < 128; ++k) a = fmaf(pedRow[k], Wih_cell[t * 384 + 256 + k], a);
    cellb[t] = a;
}

// veh_agg = sumVW - gvh @ w_graph_veh -> bf16 into cellAct[:,128:256]
__global__ __launch_bounds__(256) void vehagg_kernel(const __bf16* __restrict__ actV,
                                                     const __bf16* __restrict__ Wpk,
                                                     const float* __restrict__ sumVW,
                                                     __bf16* cellAct) {
    const int lane = threadIdx.x & 31;
    const int wave = threadIdx.x >> 5;
    const int row0 = blockIdx.x * 16;
    const int colBase = wave * 16;
    const int m = lane & 15;
    v8f acc = {};
#pragma unroll
    for (int kc = 0; kc < 4; ++kc) {
        v16bf a = load_a_frag(actV + (size_t)(row0 + m) * 256 + 128 + kc * 32, lane);
        const v16bf* bp = reinterpret_cast<const v16bf*>(Wpk) +
                          ((size_t)(kc * 8 + wave) * 32 + lane);
        acc = WMMA_BF16(a, *bp, acc);
    }
    const int n  = lane & 15;
    const int mh = (lane & 16) ? 8 : 0;
    const float s = sumVW[colBase + n];
#pragma unroll
    for (int r = 0; r < 8; ++r)
        cellAct[(size_t)(row0 + mh + r) * 512 + 128 + colBase + n] = (__bf16)(s - acc[r]);
}

// ---------------------------------------------------------------------------
// Cell LSTM: A = [dyn | veh_agg | ch] bf16 [N,512]; B packed [512 -> 1024].
// ped_agg contribution pre-folded into cellb. Fused gates + output projection.
// Block: 16 rows x 1024 cols, 8 waves x 8 tiles, 64 KB dynamic LDS.
// ---------------------------------------------------------------------------
__global__ __launch_bounds__(256) void cell_kernel(
    __bf16* cellAct, const __bf16* __restrict__ Wpk, const float* __restrict__ cellb,
    float* ch, float* cc,
    const float* __restrict__ W_out, const float* __restrict__ b_out,
    float* out, int s) {
    extern __shared__ float g[];  // [16][1024]
    const int lane = threadIdx.x & 31;
    const int wave = threadIdx.x >> 5;
    const int row0 = blockIdx.x * 16;
    const int colBase = wave * 128;
    const int m = lane & 15;
    v8f acc[8] = {};
    for (int kc = 0; kc < 16; ++kc) {
        v16bf a = load_a_frag(cellAct + (size_t)(row0 + m) * 512 + kc * 32, lane);
        const v16bf* bp = reinterpret_cast<const v16bf*>(Wpk) +
                          ((size_t)(kc * 64 + (colBase >> 4)) * 32 + lane);
#pragma unroll
        for (int t = 0; t < 8; ++t) acc[t] = WMMA_BF16(a, bp[t * 32], acc[t]);
    }
    const int n  = lane & 15;
    const int mh = (lane & 16) ? 8 : 0;
#pragma unroll
    for (int t = 0; t < 8; ++t)
#pragma unroll
        for (int r = 0; r < 8; ++r) {
            int col = colBase + t * 16 + n;
            g[(mh + r) * 1024 + col] = acc[t][r] + cellb[col];
        }
    __syncthreads();
    // gates (H=256), stash new h back into g[r][0:256] for the output projection
    for (int e = threadIdx.x; e < 16 * 256; e += 256) {
        int r = e >> 8, h = e & 255, node = row0 + r;
        float gi = g[r * 1024 + h],       gf = g[r * 1024 + 256 + h];
        float gg = g[r * 1024 + 512 + h], go = g[r * 1024 + 768 + h];
        float cp = cc[(size_t)node * 256 + h];
        float c2 = sigm(gf) * cp + sigm(gi) * tanhf(gg);
        float h2 = sigm(go) * tanhf(c2);
        cc[(size_t)node * 256 + h] = c2;
        ch[(size_t)node * 256 + h] = h2;
        cellAct[(size_t)node * 512 + 256 + h] = (__bf16)h2;  // A-panel for next step
        g[r * 1024 + h] = h2;
    }
    __syncthreads();
    if (threadIdx.x < 80) {  // 16 rows x 5 outputs
        int r = threadIdx.x / 5, j = threadIdx.x % 5;
        float a = b_out[j];
        for (int k = 0; k < 256; ++k) a = fmaf(g[r * 1024 + k], W_out[j * 256 + k], a);
        out[((size_t)s * NN + row0 + r) * 5 + j] = a;
    }
}

// Final carry copy into d_out (after the S*N*OUT outputs block)
__global__ void fin_kernel(const float* ch, const float* cc,
                           const float* gvh, const float* gvc,
                           const float* gph, const float* gpc, float* out) {
    const size_t base = (size_t)SEQ * NN * 5;
    for (int t = blockIdx.x * blockDim.x + threadIdx.x; t < NN * 256;
         t += gridDim.x * blockDim.x) {
        out[base + t] = ch[t];
        out[base + (size_t)NN * 256 + t] = cc[t];
        if (t < NN * 128) {
            const size_t b2 = base + (size_t)2 * NN * 256;
            out[b2 + t] = gvh[t];
            out[b2 + (size_t)NN * 128 + t] = gvc[t];
            out[b2 + (size_t)2 * NN * 128 + t] = gph[t];
            out[b2 + (size_t)3 * NN * 128 + t] = gpc[t];
        }
    }
}

// ---------------------------------------------------------------------------
extern "C" void kernel_launch(void* const* d_in, const int* in_sizes, int n_in,
                              void* d_out, int out_size, void* d_ws, size_t ws_size,
                              hipStream_t stream) {
    const float* veh_in   = (const float*)d_in[0];
    const float* ped_in   = (const float*)d_in[1];
    const float* cell_h0  = (const float*)d_in[2];
    const float* cell_c0  = (const float*)d_in[3];
    const float* gveh_h0  = (const float*)d_in[4];
    const float* gveh_c0  = (const float*)d_in[5];
    const float* gped_h0  = (const float*)d_in[6];
    const float* gped_c0  = (const float*)d_in[7];
    const float* W_dyn    = (const float*)d_in[8];
    const float* b_dyn    = (const float*)d_in[9];
    const float* W_gve    = (const float*)d_in[10];
    const float* b_gve    = (const float*)d_in[11];
    const float* W_gpe    = (const float*)d_in[12];
    const float* b_gpe    = (const float*)d_in[13];
    const float* Wih_gveh = (const float*)d_in[14];
    const float* bih_gveh = (const float*)d_in[15];
    const float* Whh_gveh = (const float*)d_in[16];
    const float* bhh_gveh = (const float*)d_in[17];
    const float* Wih_gped = (const float*)d_in[18];
    const float* bih_gped = (const float*)d_in[19];
    const float* Whh_gped = (const float*)d_in[20];
    const float* bhh_gped = (const float*)d_in[21];
    const float* Wih_cell = (const float*)d_in[22];
    const float* bih_cell = (const float*)d_in[23];
    const float* Whh_cell = (const float*)d_in[24];
    const float* bhh_cell = (const float*)d_in[25];
    const float* w_gveh   = (const float*)d_in[26];
    const float* w_gped   = (const float*)d_in[27];
    const float* W_out    = (const float*)d_in[28];
    const float* b_out    = (const float*)d_in[29];
    float* out = (float*)d_out;

    // ---- workspace carve-up (~26 MB) ----
    char* ws = (char*)d_ws;
    size_t off = 0;
    auto alloc = [&](size_t bytes) -> void* {
        void* p = ws + off;
        off = (off + bytes + 255) & ~(size_t)255;
        return p;
    };
    __bf16* gvehWpk = (__bf16*)alloc(256 * 512 * 2);
    __bf16* gpedWpk = (__bf16*)alloc(256 * 512 * 2);
    __bf16* cellWpk = (__bf16*)alloc(512 * 1024 * 2);
    __bf16* wgvPk   = (__bf16*)alloc(128 * 128 * 2);
    float* bV    = (float*)alloc(512 * 4);
    float* bP    = (float*)alloc(512 * 4);
    float* cellb = (float*)alloc(1024 * 4);
    float* sumV  = (float*)alloc(128 * 4);
    float* sumP  = (float*)alloc(128 * 4);
    float* sumVW = (float*)alloc(128 * 4);
    __bf16* actV    = (__bf16*)alloc((size_t)NN * 256 * 2);
    __bf16* actP    = (__bf16*)alloc((size_t)NN * 256 * 2);
    __bf16* cellAct = (__bf16*)alloc((size_t)NN * 512 * 2);
    float* gvh = (float*)alloc((size_t)NN * 128 * 4);
    float* gvc = (float*)alloc((size_t)NN * 128 * 4);
    float* gph = (float*)alloc((size_t)NN * 128 * 4);
    float* gpc = (float*)alloc((size_t)NN * 128 * 4);
    float* ch  = (float*)alloc((size_t)NN * 256 * 4);
    float* cc  = (float*)alloc((size_t)NN * 256 * 4);
    (void)ws_size; (void)in_sizes; (void)n_in; (void)out_size;

    // ---- weight prep (every call; deterministic) ----
    pack_b_kernel<<<256, 256, 0, stream>>>(gvehWpk, Wih_gveh, Whh_gveh, 256, 128, 512, 128, 128, 0);
    pack_b_kernel<<<256, 256, 0, stream>>>(gpedWpk, Wih_gped, Whh_gped, 256, 128, 512, 128, 128, 0);
    pack_b_kernel<<<512, 256, 0, stream>>>(cellWpk, Wih_cell, Whh_cell, 512, 256, 1024, 384, 256, 0);
    pack_b_kernel<<<64, 256, 0, stream>>>(wgvPk, w_gveh, nullptr, 128, 128, 128, 128, 128, 1);
    bias_kernel<<<2, 256, 0, stream>>>(bih_gveh, bhh_gveh, bih_gped, bhh_gped, bV, bP);
    init_kernel<<<1024, 256, 0, stream>>>(cell_h0, cell_c0, gveh_h0, gveh_c0, gped_h0, gped_c0,
                                          ch, cc, gvh, gvc, gph, gpc, cellAct, actV, actP);

    // ---- sequential scan ----
    for (int s = 0; s < SEQ; ++s) {
        const float* veh = veh_in + (size_t)s * NN * 2;
        const float* ped = ped_in + (size_t)s * NN * 2;
        embed_kernel<<<1024, 256, 0, stream>>>(veh, ped, W_dyn, b_dyn, W_gve, b_gve,
                                               W_gpe, b_gpe, cellAct, actV, actP);
        glstm_kernel<<<dim3(NN / 16, 2), 256, 0, stream>>>(actV, actP, gvehWpk, gpedWpk,
                                                           bV, bP, gvh, gvc, gph, gpc);
        reduce_kernel<<<dim3(128, 2), 256, 0, stream>>>(gvh, gph, sumV, sumP);
        proj_kernel<<<1, 1024, 0, stream>>>(sumV, sumP, w_gveh, w_gped, Wih_cell,
                                            bih_cell, bhh_cell, sumVW, cellb);
        vehagg_kernel<<<NN / 16, 256, 0, stream>>>(actV, wgvPk, sumVW, cellAct);
        cell_kernel<<<NN / 16, 256, 16 * 1024 * sizeof(float), stream>>>(
            cellAct, cellWpk, cellb, ch, cc, W_out, b_out, out, s);
    }
    fin_kernel<<<1024, 256, 0, stream>>>(ch, cc, gvh, gvc, gph, gpc, out);
}
